// LlamaAttention_52269751992821
// MI455X (gfx1250) — matrix operand order repaired and checked by reading, not compiled, not run on hardware
//
#include <hip/hip_runtime.h>
#include <cstdint>

// ---------------------------------------------------------------------------
// LlamaAttention forward for MI455X (gfx1250), wave32 + WMMA bf16 (hi/lo split)
// Round 2: conversions hoisted out of hot loops (bf16 LDS tiles / bf16 q,k,v)
// ---------------------------------------------------------------------------

typedef __attribute__((ext_vector_type(16))) __bf16 v16bf;
typedef __attribute__((ext_vector_type(8)))  float  v8f;

#define S_LEN 2048
#define HID_D 2048
#define NH    32
#define NKV   8
#define HD    64
#define SCALE 0.125f   // 64^-0.5
#define NEGBIG (-1e30f)
#define LDA_P 40       // padded LDS row stride in bf16 (80 B: 16B-aligned, 20-bank)

static __device__ __forceinline__ void wmma_acc(v8f& c, const v16bf& a, const v16bf& b) {
    // D = A*B + C ; f32 accumulate, bf16 inputs (V_WMMA_F32_16X16X32_BF16)
    c = __builtin_amdgcn_wmma_f32_16x16x32_bf16(false, a, false, b, (short)0, c, false, false);
}

struct bfp { __bf16 hi; __bf16 lo; };
static __device__ __forceinline__ bfp split2(float f) {
    bfp r;
    r.hi = (__bf16)f;                 // RNE
    r.lo = (__bf16)(f - (float)r.hi); // residual
    return r;
}

// ---------------------------------------------------------------------------
// Tiled GEMM:  C[M x N] = A[M x K] @ B[K x N], all f32 row-major.
// bf16x3: C += Ahi*Bhi + Ahi*Blo + Alo*Bhi  (~fp32 accuracy).
// A/B staged in LDS as bf16 hi/lo (convert-once); B staged transposed so all
// fragment reads are contiguous 16B-aligned runs (ds_load_b128-mergeable).
// Block 128x128, 8 waves (4x2), wave tile 32x64 (2x4 WMMA subtiles).
// storeMode 0: row-major (ldc = N). storeMode 1: head-major [col/64][row][64].
// ---------------------------------------------------------------------------
__global__ __launch_bounds__(256)
void gemm_bf16x3(const float* __restrict__ A, const float* __restrict__ B,
                 float* __restrict__ C, int M, int N, int K, int storeMode) {
    __shared__ __bf16 AsH[128 * LDA_P];   // [row][k]   10 KB
    __shared__ __bf16 AsL[128 * LDA_P];
    __shared__ __bf16 BsH[128 * LDA_P];   // [col][k] (transposed)
    __shared__ __bf16 BsL[128 * LDA_P];

    const int t    = threadIdx.x;
    const int lane = t & 31;
    const int wid  = t >> 5;
    const int half = lane >> 4;
    const int ln   = lane & 15;
    const int bm0  = blockIdx.y * 128;
    const int bn0  = blockIdx.x * 128;
    const int wRow = (wid >> 1) * 32;
    const int wCol = (wid & 1) * 64;

    v8f acc[2][4];
    for (int i = 0; i < 2; ++i)
        for (int j = 0; j < 4; ++j)
            for (int r = 0; r < 8; ++r) acc[i][j][r] = 0.f;

    for (int k0 = 0; k0 < K; k0 += 32) {
        // ---- stage A tile 128x32 -> bf16 hi/lo ----
        for (int i = 0; i < 4; ++i) {
            int idx = i * 256 + t;
            int r = idx >> 3, q4 = (idx & 7) * 4;
            const float4 d = *(const float4*)(A + (size_t)(bm0 + r) * K + k0 + q4);
            float vv[4] = {d.x, d.y, d.z, d.w};
            __bf16* sh = &AsH[r * LDA_P + q4];
            __bf16* sl = &AsL[r * LDA_P + q4];
            for (int e = 0; e < 4; ++e) { bfp p = split2(vv[e]); sh[e] = p.hi; sl[e] = p.lo; }
        }
        // ---- stage B tile 32x128 -> bf16 hi/lo, transposed [col][k] ----
        for (int i = 0; i < 4; ++i) {
            int idx = i * 256 + t;
            int r = idx >> 5, q4 = (idx & 31) * 4;
            const float4 d = *(const float4*)(B + (size_t)(k0 + r) * N + bn0 + q4);
            float vv[4] = {d.x, d.y, d.z, d.w};
            for (int e = 0; e < 4; ++e) {
                bfp p = split2(vv[e]);
                BsH[(q4 + e) * LDA_P + r] = p.hi;
                BsL[(q4 + e) * LDA_P + r] = p.lo;
            }
        }
        if (k0 + 32 < K) { // prefetch next K-slab (global_prefetch_b8)
            __builtin_prefetch(A + (size_t)(bm0 + (t >> 3)) * K + (k0 + 32) + (t & 7) * 4, 0, 0);
            __builtin_prefetch(B + (size_t)(k0 + 32 + (t >> 5)) * N + bn0 + (t & 31) * 4, 0, 0);
        }
        __syncthreads();

        // ---- fragments: contiguous bf16 runs, no conversion ----
        v16bf aH[2], aL[2], bH[4], bL[4];
        for (int i = 0; i < 2; ++i) {
            const __bf16* ph = &AsH[(wRow + 16 * i + ln) * LDA_P];
            const __bf16* pl = &AsL[(wRow + 16 * i + ln) * LDA_P];
            for (int e = 0; e < 8; ++e) { aH[i][e]     = ph[16 * half + e];     aL[i][e]     = pl[16 * half + e]; }
            for (int e = 0; e < 8; ++e) { aH[i][8 + e] = ph[16 * half + 8 + e]; aL[i][8 + e] = pl[16 * half + 8 + e]; }
        }
        for (int j = 0; j < 4; ++j) {
            const int col = wCol + 16 * j + ln;
            const __bf16* ph = &BsH[col * LDA_P + 16 * half];
            const __bf16* pl = &BsL[col * LDA_P + 16 * half];
            for (int e = 0; e < 16; ++e) { bH[j][e] = ph[e]; bL[j][e] = pl[e]; }
        }
        for (int i = 0; i < 2; ++i)
            for (int j = 0; j < 4; ++j) {
                wmma_acc(acc[i][j], aH[i], bH[j]);
                wmma_acc(acc[i][j], aH[i], bL[j]);
                wmma_acc(acc[i][j], aL[i], bH[j]);
            }
        __syncthreads();
    }

    // ---- store ----
    for (int i = 0; i < 2; ++i)
        for (int j = 0; j < 4; ++j)
            for (int r = 0; r < 8; ++r) {
                int row = bm0 + wRow + 16 * i + r + 8 * half;
                int col = bn0 + wCol + 16 * j + ln;
                float v = acc[i][j][r];
                if (storeMode == 0) {
                    C[(size_t)row * N + col] = v;
                } else {
                    int g = col >> 6, d = col & 63;
                    C[(size_t)g * ((size_t)M * 64) + (size_t)row * 64 + d] = v;
                }
            }
}

// ---------------------------------------------------------------------------
// RoPE + bf16 hi/lo conversion. Reads f32 head-major q [NH][S][64],
// k [NKV][S][64], v [NKV][S][64]; writes bf16 hi/lo copies (q,k rotated,
// v passthrough). Thread owns the (d, d+32) pair.
// ---------------------------------------------------------------------------
__global__ __launch_bounds__(256)
void rope_cvt_kernel(const float* __restrict__ qf, const float* __restrict__ kf,
                     const float* __restrict__ vf,
                     const float* __restrict__ cost, const float* __restrict__ sint,
                     __bf16* __restrict__ qH, __bf16* __restrict__ qL,
                     __bf16* __restrict__ kH, __bf16* __restrict__ kL,
                     __bf16* __restrict__ vH, __bf16* __restrict__ vL) {
    int idx = blockIdx.x * 256 + threadIdx.x;   // (NH+2*NKV)*S*32 threads
    int d = idx & 31;
    int s = (idx >> 5) & (S_LEN - 1);
    int g = idx >> 16;                          // 32*2048 = 65536 per head
    const float* src;
    __bf16 *dH, *dL;
    bool rot;
    if (g < NH) {
        size_t off = (size_t)g * S_LEN * 64;
        src = qf + off; dH = qH + off; dL = qL + off; rot = true;
    } else if (g < NH + NKV) {
        size_t off = (size_t)(g - NH) * S_LEN * 64;
        src = kf + off; dH = kH + off; dL = kL + off; rot = true;
    } else {
        size_t off = (size_t)(g - NH - NKV) * S_LEN * 64;
        src = vf + off; dH = vH + off; dL = vL + off; rot = false;
    }
    float x1 = src[(size_t)s * 64 + d];
    float x2 = src[(size_t)s * 64 + d + 32];
    float y1 = x1, y2 = x2;
    if (rot) {
        float c1 = cost[s * 64 + d],      s1 = sint[s * 64 + d];
        float c2 = cost[s * 64 + d + 32], s2 = sint[s * 64 + d + 32];
        y1 = x1 * c1 - x2 * s1;
        y2 = x2 * c2 + x1 * s2;
    }
    bfp p1 = split2(y1), p2 = split2(y2);
    dH[(size_t)s * 64 + d]      = p1.hi;  dL[(size_t)s * 64 + d]      = p1.lo;
    dH[(size_t)s * 64 + d + 32] = p2.hi;  dL[(size_t)s * 64 + d + 32] = p2.lo;
}

// ---------------------------------------------------------------------------
// Causal attention with full-weights output. q/k/v arrive pre-split as bf16
// hi/lo (L2-resident), so score/attn fragment loads need NO conversion.
// Block = (128 q-rows, head); wave owns 16 q-rows. Two-pass online softmax;
// pass 2 recomputes scores, writes normalized weights (512 MB stream) and
// accumulates attn += w @ V via WMMA (w transposed through per-wave LDS tile).
// No block barriers (per-wave LDS regions; same-wave DS ops are in-order).
// ---------------------------------------------------------------------------
__global__ __launch_bounds__(256)
void attn_kernel(const __bf16* __restrict__ qH, const __bf16* __restrict__ qL,
                 const __bf16* __restrict__ kHg, const __bf16* __restrict__ kLg,
                 const __bf16* __restrict__ vHg, const __bf16* __restrict__ vLg,
                 float* __restrict__ wout, float* __restrict__ attn) {
    __shared__ float wlds[8][16 * 33];

    const int h    = blockIdx.y;
    const int lane = threadIdx.x & 31;
    const int wid  = threadIdx.x >> 5;
    const int half = lane >> 4;
    const int ln   = lane & 15;
    const int q0   = blockIdx.x * 128 + wid * 16;
    const int kvh  = h >> 2;                    // NH/NKV = 4

    const __bf16* kH = kHg + (size_t)kvh * S_LEN * 64;
    const __bf16* kL = kLg + (size_t)kvh * S_LEN * 64;
    const __bf16* vH = vHg + (size_t)kvh * S_LEN * 64;
    const __bf16* vL = vLg + (size_t)kvh * S_LEN * 64;

    // q strip (16 x 64) as resident A-fragments, two K=32 chunks, hi/lo
    v16bf qfH[2], qfL[2];
    {
        const __bf16* bh = qH + (size_t)h * S_LEN * 64 + (size_t)(q0 + ln) * 64;
        const __bf16* bl = qL + (size_t)h * S_LEN * 64 + (size_t)(q0 + ln) * 64;
        for (int c = 0; c < 2; ++c) {
            for (int e = 0; e < 8; ++e) { qfH[c][e]     = bh[32 * c + 16 * half + e];     qfL[c][e]     = bl[32 * c + 16 * half + e]; }
            for (int e = 0; e < 8; ++e) { qfH[c][8 + e] = bh[32 * c + 16 * half + 8 + e]; qfL[c][8 + e] = bl[32 * c + 16 * half + 8 + e]; }
        }
    }

    float mrow[8], lrow[8];
    for (int r = 0; r < 8; ++r) { mrow[r] = NEGBIG; lrow[r] = 0.f; }

    const int nAct = (q0 + 15) / 32 + 1;        // causal: active 32-key chunks

    // ---------------- pass 1: row max + sumexp ----------------
    for (int ch = 0; ch < nAct; ++ch) {
        const int k0 = ch * 32;
        v8f c0, c1;
        for (int r = 0; r < 8; ++r) { c0[r] = 0.f; c1[r] = 0.f; }
        for (int kc = 0; kc < 2; ++kc) {
            v16bf bH0, bL0, bH1, bL1;
            {
                const size_t o = (size_t)(k0 + ln) * 64 + 32 * kc + 16 * half;
                for (int e = 0; e < 16; ++e) { bH0[e] = kH[o + e]; bL0[e] = kL[o + e]; }
            }
            {
                const size_t o = (size_t)(k0 + 16 + ln) * 64 + 32 * kc + 16 * half;
                for (int e = 0; e < 16; ++e) { bH1[e] = kH[o + e]; bL1[e] = kL[o + e]; }
            }
            wmma_acc(c0, qfH[kc], bH0); wmma_acc(c0, qfH[kc], bL0); wmma_acc(c0, qfL[kc], bH0);
            wmma_acc(c1, qfH[kc], bH1); wmma_acc(c1, qfH[kc], bL1); wmma_acc(c1, qfL[kc], bH1);
        }
        for (int r = 0; r < 8; ++r) {
            int qi = q0 + r + 8 * half;
            float s0 = (k0 + ln      <= qi) ? c0[r] * SCALE : NEGBIG;
            float s1 = (k0 + 16 + ln <= qi) ? c1[r] * SCALE : NEGBIG;
            float mx = fmaxf(s0, s1);
            for (int msk = 8; msk >= 1; msk >>= 1) mx = fmaxf(mx, __shfl_xor(mx, msk, 32));
            float mnew = fmaxf(mrow[r], mx);
            float e = __expf(s0 - mnew) + __expf(s1 - mnew);
            for (int msk = 8; msk >= 1; msk >>= 1) e += __shfl_xor(e, msk, 32);
            lrow[r] = lrow[r] * __expf(mrow[r] - mnew) + e;
            mrow[r] = mnew;
        }
    }

    float inv_l[8];
    for (int r = 0; r < 8; ++r) inv_l[r] = 1.0f / lrow[r];

    v8f accD[4];
    for (int jt = 0; jt < 4; ++jt)
        for (int r = 0; r < 8; ++r) accD[jt][r] = 0.f;

    float* wl = wlds[wid];                       // per-wave private LDS region

    // ---------------- pass 2: weights out + attn accumulate ----------------
    for (int ch = 0; ch < S_LEN / 32; ++ch) {
        const int k0 = ch * 32;
        if (ch < nAct) {
            v8f c0, c1;
            for (int r = 0; r < 8; ++r) { c0[r] = 0.f; c1[r] = 0.f; }
            for (int kc = 0; kc < 2; ++kc) {
                v16bf bH0, bL0, bH1, bL1;
                {
                    const size_t o = (size_t)(k0 + ln) * 64 + 32 * kc + 16 * half;
                    for (int e = 0; e < 16; ++e) { bH0[e] = kH[o + e]; bL0[e] = kL[o + e]; }
                }
                {
                    const size_t o = (size_t)(k0 + 16 + ln) * 64 + 32 * kc + 16 * half;
                    for (int e = 0; e < 16; ++e) { bH1[e] = kH[o + e]; bL1[e] = kL[o + e]; }
                }
                wmma_acc(c0, qfH[kc], bH0); wmma_acc(c0, qfH[kc], bL0); wmma_acc(c0, qfL[kc], bH0);
                wmma_acc(c1, qfH[kc], bH1); wmma_acc(c1, qfH[kc], bL1); wmma_acc(c1, qfL[kc], bH1);
            }
            // normalized weights -> global + LDS stage (C-layout -> A-layout)
            for (int r = 0; r < 8; ++r) {
                int qi = q0 + r + 8 * half;
                float w0 = (k0 + ln      <= qi) ? __expf(c0[r] * SCALE - mrow[r]) * inv_l[r] : 0.f;
                float w1 = (k0 + 16 + ln <= qi) ? __expf(c1[r] * SCALE - mrow[r]) * inv_l[r] : 0.f;
                size_t o = ((size_t)h * S_LEN + qi) * S_LEN + k0 + ln;
                wout[o]      = w0;
                wout[o + 16] = w1;
                wl[(r + 8 * half) * 33 + ln]      = w0;
                wl[(r + 8 * half) * 33 + 16 + ln] = w1;
            }
            // w as A-fragment (16x32, K = keys), hi/lo (fresh values -> split here)
            v16bf wfH, wfL;
            {
                const float* base = &wl[ln * 33];
                for (int e = 0; e < 8; ++e) { bfp p = split2(base[8 * half + e]);      wfH[e]     = p.hi; wfL[e]     = p.lo; }
                for (int e = 0; e < 8; ++e) { bfp p = split2(base[16 + 8 * half + e]); wfH[8 + e] = p.hi; wfL[8 + e] = p.lo; }
            }
            // attn(16x64) += w(16x32) @ V(32x64) over 4 d-tiles
            for (int jt = 0; jt < 4; ++jt) {
                v16bf bvH, bvL;
                for (int e = 0; e < 16; ++e) {
                    size_t o = (size_t)(k0 + 8 * half + e + (e >= 8 ? 8 : 0)) * 64 + jt * 16 + ln;
                    bvH[e] = vH[o]; bvL[e] = vL[o];
                }
                wmma_acc(accD[jt], wfH, bvH);
                wmma_acc(accD[jt], wfH, bvL);
                wmma_acc(accD[jt], wfL, bvH);
            }
        } else {
            // fully-masked chunk: weights are exactly zero
            for (int r = 0; r < 8; ++r) {
                int qi = q0 + r + 8 * half;
                size_t o = ((size_t)h * S_LEN + qi) * S_LEN + k0 + ln;
                wout[o]      = 0.f;
                wout[o + 16] = 0.f;
            }
        }
    }

    // attn -> [S][NH*HD] for the Wo GEMM
    for (int jt = 0; jt < 4; ++jt)
        for (int r = 0; r < 8; ++r) {
            int qi = q0 + r + 8 * half;
            attn[(size_t)qi * (NH * HD) + h * 64 + jt * 16 + ln] = accD[jt][r];
        }
}

// ---------------------------------------------------------------------------
extern "C" void kernel_launch(void* const* d_in, const int* in_sizes, int n_in,
                              void* d_out, int out_size, void* d_ws, size_t ws_size,
                              hipStream_t stream) {
    const float* hidden = (const float*)d_in[0];
    const float* cost   = (const float*)d_in[1];
    const float* sint   = (const float*)d_in[2];
    const float* Wq     = (const float*)d_in[3];
    const float* Wk     = (const float*)d_in[4];
    const float* Wv     = (const float*)d_in[5];
    const float* Wo     = (const float*)d_in[6];

    float* out     = (float*)d_out;                       // [S, HID]
    float* weights = out + (size_t)S_LEN * HID_D;         // [NH, S, S]

    const size_t QE = (size_t)NH  * S_LEN * HD;           // 4M elements
    const size_t KE = (size_t)NKV * S_LEN * HD;           // 1M elements

    float* wsf  = (float*)d_ws;
    float* qf   = wsf;                                    // f32 q   16 MB
    float* kf   = qf + QE;                                // f32 k    4 MB
    float* vf   = kf + KE;                                // f32 v    4 MB
    float* abuf = vf + KE;                                // f32 attn 16 MB
    __bf16* bb  = (__bf16*)(abuf + QE);
    __bf16* qHb = bb;            __bf16* qLb = qHb + QE;  // bf16 q hi/lo 16 MB
    __bf16* kHb = qLb + QE;      __bf16* kLb = kHb + KE;  // bf16 k hi/lo  4 MB
    __bf16* vHb = kLb + KE;      __bf16* vLb = vHb + KE;  // bf16 v hi/lo  4 MB

    dim3 blk(256);
    // QKV projections (head-major f32 outputs)
    gemm_bf16x3<<<dim3(HID_D / 128, S_LEN / 128), blk, 0, stream>>>(hidden, Wq, qf, S_LEN, NH  * HD, HID_D, 1);
    gemm_bf16x3<<<dim3((NKV * HD) / 128, S_LEN / 128), blk, 0, stream>>>(hidden, Wk, kf, S_LEN, NKV * HD, HID_D, 1);
    gemm_bf16x3<<<dim3((NKV * HD) / 128, S_LEN / 128), blk, 0, stream>>>(hidden, Wv, vf, S_LEN, NKV * HD, HID_D, 1);
    // RoPE + hi/lo bf16 conversion of q, k, v
    rope_cvt_kernel<<<dim3(((NH + 2 * NKV) * S_LEN * 32) / 256), blk, 0, stream>>>(
        qf, kf, vf, cost, sint, qHb, qLb, kHb, kLb, vHb, vLb);
    // Attention (+ full weights output)
    attn_kernel<<<dim3(S_LEN / 128, NH), blk, 0, stream>>>(qHb, qLb, kHb, kLb, vHb, vLb, weights, abuf);
    // Output projection
    gemm_bf16x3<<<dim3(HID_D / 128, S_LEN / 128), blk, 0, stream>>>(abuf, Wo, out, S_LEN, HID_D, NH * HD, 0);
}
